// GAT_2379411882410
// MI455X (gfx1250) — matrix operand order repaired and checked by reading, not compiled, not run on hardware
//
#include <hip/hip_runtime.h>
#include <hip/hip_bf16.h>

// ---------------------------------------------------------------------------
// 2-layer GAT for MI455X (gfx1250, wave32).
// GEMMs: v_wmma_f32_16x16x32_bf16, A panel staged once in LDS (single barrier),
// fully unrolled K loop, 2 N-tiles per wave for A-fragment reuse.
// Edge softmax/aggregation: f32 atomics (L2-resident working set).
// ---------------------------------------------------------------------------

typedef __bf16 bf16_t;
typedef bf16_t v16bf __attribute__((ext_vector_type(16)));
typedef float  v8f   __attribute__((ext_vector_type(8)));

#define NNODES   50000
#define NEDGES   400000
#define INFEAT   256
#define NHEADS   8
#define HIDF     32
#define HHID     256   // NHEADS*HIDF
#define OUTF     64
#define NEGSLOPE 0.2f
#define GEMMK    256   // K dim of both layer GEMMs

// ---------------------------- utility kernels ------------------------------

__global__ void k_fill_f32(float* p, float v, int n) {
    int i = blockIdx.x * blockDim.x + threadIdx.x;
    if (i < n) p[i] = v;
}

__global__ void k_cvt_bf16(const float* __restrict__ s, bf16_t* __restrict__ d, int n) {
    int i = blockIdx.x * blockDim.x + threadIdx.x;
    if (i < n) d[i] = (bf16_t)s[i];
}

// W [K,N] row-major f32  ->  WT [N,K] row-major bf16
__global__ void k_transpose_cvt(const float* __restrict__ w, bf16_t* __restrict__ wt,
                                int K, int N) {
    int i = blockIdx.x * blockDim.x + threadIdx.x;
    if (i >= K * N) return;
    int k = i / N, n = i % N;
    wt[(size_t)n * K + k] = (bf16_t)w[i];
}

// ------------------------------ WMMA GEMM ----------------------------------
// C[M,Nc] = A[M,256] (bf16, row-major) * BT[Nc,256]^T (bf16, N-major); C f32.
// Block: 256 threads = 8 waves. Block owns a 16-row M panel (staged in LDS
// once) and up to 256 output columns; each wave computes two 16x16 C tiles.

__device__ inline v16bf load_fragA_lds(const bf16_t* a, int ld) {
    // 16-bit A 16x32: lanes 0-15 -> K 0..15, lanes 16-31 -> K 16..31
    int lane = threadIdx.x & 31;
    int m  = lane & 15;
    int kh = (lane >> 4) * 16;
    const bf16_t* p = a + m * ld + kh;
    v16bf r;
#pragma unroll
    for (int i = 0; i < 16; ++i) r[i] = p[i];
    return r;
}

__device__ inline v16bf load_fragB_glb(const bf16_t* bt, int ldk) {
    // 16-bit B 32x16 (contiguous k, same as A): lanes 0-15 -> K 0..15, lanes 16-31 -> K 16..31
    int lane = threadIdx.x & 31;
    int n  = lane & 15;
    int kh = (lane >> 4) * 16;
    const bf16_t* p = bt + (size_t)n * ldk + kh;
    v16bf r;
#pragma unroll
    for (int i = 0; i < 16; ++i) r[i] = p[i];
    return r;
}

__global__ void __launch_bounds__(256)
k_gemm_bf16_wmma(const bf16_t* __restrict__ A, const bf16_t* __restrict__ BT,
                 float* __restrict__ C, int Nc) {
    __shared__ __attribute__((aligned(16))) bf16_t sA[16 * GEMMK];   // 8 KB

    const int wid = threadIdx.x >> 5;
    const int m0  = blockIdx.x * 16;

    // two N tiles per wave; clamp -> redundant identical work, never OOB
    int nt0 = blockIdx.y * 16 + wid * 2;
    const int ntClamp = Nc / 16 - 2;
    if (nt0 > ntClamp) nt0 = ntClamp;
    const bf16_t* bt0 = BT + (size_t)nt0 * 16 * GEMMK;
    const bf16_t* bt1 = bt0 + (size_t)16 * GEMMK;

    // --- stage full 16x256 A panel: 32 B per thread (2x uint4), one barrier ---
    {
        const int row = threadIdx.x >> 4;          // 16 threads per row
        const int col = (threadIdx.x & 15) * 16;   // 16 bf16 = 32 B per thread
        const uint4* ag = (const uint4*)(A + (size_t)(m0 + row) * GEMMK + col);
        uint4* as = (uint4*)(sA + row * GEMMK + col);
        as[0] = ag[0];
        as[1] = ag[1];
        __builtin_prefetch(A + (size_t)(m0 + 16 + row) * GEMMK + col, 0, 1);
    }
    __syncthreads();

    v8f acc0 = {}, acc1 = {};
#pragma unroll
    for (int k0 = 0; k0 < GEMMK; k0 += 32) {
        v16bf fa  = load_fragA_lds(sA + k0, GEMMK);
        v16bf fb0 = load_fragB_glb(bt0 + k0, GEMMK);
        v16bf fb1 = load_fragB_glb(bt1 + k0, GEMMK);
        acc0 = __builtin_amdgcn_wmma_f32_16x16x32_bf16(
                   false, fa, false, fb0, (short)0, acc0, false, false);
        acc1 = __builtin_amdgcn_wmma_f32_16x16x32_bf16(
                   false, fa, false, fb1, (short)0, acc1, false, false);
    }

    const int lane = threadIdx.x & 31;
    const int n0   = nt0 * 16 + (lane & 15);
    const int mb   = m0 + ((lane >> 4) * 8);
#pragma unroll
    for (int r = 0; r < 8; ++r) {
        C[(size_t)(mb + r) * Nc + n0]      = acc0[r];
        C[(size_t)(mb + r) * Nc + n0 + 16] = acc1[r];
    }
}

// --------------------------- attention kernels -----------------------------

// es[n,h] = sum_f h[n,h,f]*a_src[h,f];  ed likewise.
__global__ void k_node_scores(const float* __restrict__ h,
                              const float* __restrict__ a_s,
                              const float* __restrict__ a_d,
                              float* __restrict__ es, float* __restrict__ ed,
                              int N, int H, int F) {
    int i = blockIdx.x * blockDim.x + threadIdx.x;
    if (i >= N * H) return;
    int n = i / H, hh = i % H;
    const float* hp = h + (size_t)n * H * F + hh * F;
    const float* as = a_s + hh * F;
    const float* ad = a_d + hh * F;
    float s = 0.f, d = 0.f;
    for (int f = 0; f < F; ++f) { s += hp[f] * as[f]; d += hp[f] * ad[f]; }
    es[i] = s; ed[i] = d;
}

__device__ inline float edge_score(const float* es, const float* ed,
                                   int s, int d, int hh, int H) {
    float v = es[s * H + hh] + ed[d * H + hh];
    return v > 0.f ? v : NEGSLOPE * v;
}

__device__ inline void atomicMaxF32(float* addr, float val) {
    unsigned int* ua = reinterpret_cast<unsigned int*>(addr);
    unsigned int old = __float_as_uint(*addr);
    while (__uint_as_float(old) < val) {
        unsigned int assumed = old;
        old = atomicCAS(ua, assumed, __float_as_uint(val));
        if (old == assumed) break;
    }
}

__global__ void k_edge_max(const float* __restrict__ es, const float* __restrict__ ed,
                           const int* __restrict__ src, const int* __restrict__ dst,
                           float* __restrict__ nmax, int E, int H) {
    int i = blockIdx.x * blockDim.x + threadIdx.x;
    if (i >= E * H) return;
    int e = i / H, hh = i % H;
    int s = src[e], d = dst[e];
    atomicMaxF32(&nmax[d * H + hh], edge_score(es, ed, s, d, hh, H));
}

__global__ void k_edge_sum(const float* __restrict__ es, const float* __restrict__ ed,
                           const int* __restrict__ src, const int* __restrict__ dst,
                           const float* __restrict__ nmax, float* __restrict__ nsum,
                           int E, int H) {
    int i = blockIdx.x * blockDim.x + threadIdx.x;
    if (i >= E * H) return;
    int e = i / H, hh = i % H;
    int s = src[e], d = dst[e];
    float v = edge_score(es, ed, s, d, hh, H);
    atomicAdd(&nsum[d * H + hh], __expf(v - nmax[d * H + hh]));
}

__global__ void k_edge_aggregate(const float* __restrict__ es, const float* __restrict__ ed,
                                 const int* __restrict__ src, const int* __restrict__ dst,
                                 const float* __restrict__ nmax, const float* __restrict__ nsum,
                                 const float* __restrict__ h, float* __restrict__ agg,
                                 int E, int H, int F) {
    int i = blockIdx.x * blockDim.x + threadIdx.x;
    if (i >= E * H) return;
    int e = i / H, hh = i % H;
    int s = src[e], d = dst[e];
    float v = edge_score(es, ed, s, d, hh, H);
    float alpha = __expf(v - nmax[d * H + hh]) / nsum[d * H + hh];
    const float* hs = h   + (size_t)s * H * F + hh * F;
    float*       ag = agg + (size_t)d * H * F + hh * F;
    for (int f = 0; f < F; ++f) atomicAdd(&ag[f], alpha * hs[f]);
}

// out = ELU(agg + b) -> bf16 (inter-layer activation)
__global__ void k_bias_elu_cvt(const float* __restrict__ agg, const float* __restrict__ b,
                               bf16_t* __restrict__ out, int n, int C) {
    int i = blockIdx.x * blockDim.x + threadIdx.x;
    if (i >= n) return;
    float v = agg[i] + b[i % C];
    out[i] = (bf16_t)(v > 0.f ? v : (__expf(v) - 1.f));
}

__global__ void k_bias_add(float* __restrict__ out, const float* __restrict__ b,
                           int n, int C) {
    int i = blockIdx.x * blockDim.x + threadIdx.x;
    if (i < n) out[i] += b[i % C];
}

// ------------------------------- launcher ----------------------------------

static inline dim3 g1(int n) { return dim3((unsigned)((n + 255) / 256)); }

extern "C" void kernel_launch(void* const* d_in, const int* in_sizes, int n_in,
                              void* d_out, int out_size, void* d_ws, size_t ws_size,
                              hipStream_t stream) {
    const float* x     = (const float*)d_in[0];
    const float* W1    = (const float*)d_in[1];
    const float* a1s   = (const float*)d_in[2];
    const float* a1d   = (const float*)d_in[3];
    const float* b1    = (const float*)d_in[4];
    const float* W2    = (const float*)d_in[5];
    const float* a2s   = (const float*)d_in[6];
    const float* a2d   = (const float*)d_in[7];
    const float* b2    = (const float*)d_in[8];
    const int*   src   = (const int*)d_in[9];
    const int*   dst   = (const int*)d_in[10];
    float*       out   = (float*)d_out;

    // ---- carve workspace (256B aligned) ----
    char* ws = (char*)d_ws;
    size_t off = 0;
    auto carve = [&](size_t bytes) {
        void* p = ws + off;
        off += (bytes + 255) & ~(size_t)255;
        return p;
    };
    bf16_t* xb    = (bf16_t*)carve((size_t)NNODES * INFEAT * 2);
    bf16_t* w1t   = (bf16_t*)carve((size_t)HHID * INFEAT * 2);   // [256(N),256(K)]
    bf16_t* w2t   = (bf16_t*)carve((size_t)OUTF * HHID * 2);     // [64(N),256(K)]
    float*  h1    = (float*) carve((size_t)NNODES * HHID * 4);
    float*  agg1  = (float*) carve((size_t)NNODES * HHID * 4);
    float*  es1   = (float*) carve((size_t)NNODES * NHEADS * 4);
    float*  ed1   = (float*) carve((size_t)NNODES * NHEADS * 4);
    float*  nmax1 = (float*) carve((size_t)NNODES * NHEADS * 4);
    float*  nsum1 = (float*) carve((size_t)NNODES * NHEADS * 4);
    bf16_t* hb    = (bf16_t*)carve((size_t)NNODES * HHID * 2);
    float*  h2    = (float*) carve((size_t)NNODES * OUTF * 4);
    float*  es2   = (float*) carve((size_t)NNODES * 4);
    float*  ed2   = (float*) carve((size_t)NNODES * 4);
    float*  nmax2 = (float*) carve((size_t)NNODES * 4);
    float*  nsum2 = (float*) carve((size_t)NNODES * 4);
    (void)ws_size; (void)n_in; (void)in_sizes; (void)out_size;

    const int NH1 = NNODES * NHEADS;
    const int EH1 = NEDGES * NHEADS;

    // ---- precision conversion ----
    k_cvt_bf16<<<g1(NNODES * INFEAT), 256, 0, stream>>>(x, xb, NNODES * INFEAT);
    k_transpose_cvt<<<g1(INFEAT * HHID), 256, 0, stream>>>(W1, w1t, INFEAT, HHID);
    k_transpose_cvt<<<g1(HHID * OUTF), 256, 0, stream>>>(W2, w2t, HHID, OUTF);

    // ---- layer 1: GEMM (WMMA bf16), 256 output cols per block ----
    k_gemm_bf16_wmma<<<dim3(NNODES / 16, 1), 256, 0, stream>>>(xb, w1t, h1, HHID);

    // ---- layer 1: attention ----
    k_node_scores<<<g1(NH1), 256, 0, stream>>>(h1, a1s, a1d, es1, ed1,
                                               NNODES, NHEADS, HIDF);
    k_fill_f32<<<g1(NH1), 256, 0, stream>>>(nmax1, -3.0e38f, NH1);
    k_fill_f32<<<g1(NH1), 256, 0, stream>>>(nsum1, 0.f, NH1);
    k_fill_f32<<<g1(NNODES * HHID), 256, 0, stream>>>(agg1, 0.f, NNODES * HHID);
    k_edge_max<<<g1(EH1), 256, 0, stream>>>(es1, ed1, src, dst, nmax1, NEDGES, NHEADS);
    k_edge_sum<<<g1(EH1), 256, 0, stream>>>(es1, ed1, src, dst, nmax1, nsum1,
                                            NEDGES, NHEADS);
    k_edge_aggregate<<<g1(EH1), 256, 0, stream>>>(es1, ed1, src, dst, nmax1, nsum1,
                                                  h1, agg1, NEDGES, NHEADS, HIDF);
    k_bias_elu_cvt<<<g1(NNODES * HHID), 256, 0, stream>>>(agg1, b1, hb,
                                                          NNODES * HHID, HHID);

    // ---- layer 2: GEMM (WMMA bf16), 64 output cols ----
    k_gemm_bf16_wmma<<<dim3(NNODES / 16, 1), 256, 0, stream>>>(hb, w2t, h2, OUTF);

    // ---- layer 2: attention (single head) ----
    k_node_scores<<<g1(NNODES), 256, 0, stream>>>(h2, a2s, a2d, es2, ed2,
                                                  NNODES, 1, OUTF);
    k_fill_f32<<<g1(NNODES), 256, 0, stream>>>(nmax2, -3.0e38f, NNODES);
    k_fill_f32<<<g1(NNODES), 256, 0, stream>>>(nsum2, 0.f, NNODES);
    k_fill_f32<<<g1(NNODES * OUTF), 256, 0, stream>>>(out, 0.f, NNODES * OUTF);
    k_edge_max<<<g1(NEDGES), 256, 0, stream>>>(es2, ed2, src, dst, nmax2, NEDGES, 1);
    k_edge_sum<<<g1(NEDGES), 256, 0, stream>>>(es2, ed2, src, dst, nmax2, nsum2,
                                               NEDGES, 1);
    k_edge_aggregate<<<g1(NEDGES), 256, 0, stream>>>(es2, ed2, src, dst, nmax2, nsum2,
                                                     h2, out, NEDGES, 1, OUTF);
    k_bias_add<<<g1(NNODES * OUTF), 256, 0, stream>>>(out, b2, NNODES * OUTF, OUTF);
}
